// PointConvNet2_33105607917646
// MI455X (gfx1250) — matrix operand order repaired and checked by reading, not compiled, not run on hardware
//
#include <hip/hip_runtime.h>

typedef __attribute__((ext_vector_type(16))) _Float16 v16h;
typedef __attribute__((ext_vector_type(8)))  _Float16 v8h;
typedef __attribute__((ext_vector_type(8)))  float    v8f;
typedef __attribute__((ext_vector_type(4)))  float    f4;

#define D_FEAT 32
#define D_IN   35
#define D_HID  64
#define D_OUT  64
#define BLOCK  256
#define WPB    8          // waves per block (wave32)

// Two-sided float max via native integer atomics (requires init = -inf).
// Positive floats compare correctly as int; negative floats compare
// inversely as uint, so min-uint implements max. Mixed cases resolve
// correctly because any non-negative float (int >= 0, uint < 0x8000'0000)
// beats any negative float (int < 0, uint >= 0x8000'0000) under both ops.
__device__ __forceinline__ void atomicMaxF32(float* addr, float v) {
  if (v >= 0.0f) atomicMax((int*)addr, __float_as_int(v));
  else           atomicMin((unsigned int*)addr, __float_as_uint(v));
}

__device__ __forceinline__ v16h cat8(v8h lo, v8h hi) {
  return __builtin_shufflevector(lo, hi, 0,1,2,3,4,5,6,7,8,9,10,11,12,13,14,15);
}

__global__ void __launch_bounds__(BLOCK)
init_out_kernel(float* __restrict__ out, int n) {
  int i = blockIdx.x * BLOCK + threadIdx.x;
  if (i < n) out[i] = -__builtin_huge_valf();
}

__global__ void __launch_bounds__(BLOCK)
pointconv_kernel(const float* __restrict__ x, const float* __restrict__ pos,
                 const long long* __restrict__ ei,
                 const float* __restrict__ W1, const float* __restrict__ b1,
                 const float* __restrict__ W2, const float* __restrict__ b2,
                 float* __restrict__ out, int nNodes, long long nEdges) {
  // Weights col-major [n][k] in f16, K padded to 64 (rows >= 35 of W1 zero).
  __shared__ __align__(16) _Float16 W1h[D_HID * 64];
  __shared__ __align__(16) _Float16 W2h[D_OUT * D_HID];
  // Per-wave staging: h tile 16x64 f16 (row-major) + 16 dst indices.
  __shared__ __align__(16) _Float16 hstage[WPB][16 * D_HID];
  __shared__ __align__(16) int      dstStage[WPB][16];

  for (int idx = threadIdx.x; idx < 64 * 64; idx += BLOCK) {
    int k = idx & 63, n = idx >> 6;
    W1h[n * 64 + k] = (k < D_IN) ? (_Float16)W1[k * D_HID + n] : (_Float16)0.0f;
    W2h[n * 64 + k] = (_Float16)W2[k * D_OUT + n];
  }
  __syncthreads();

  const int lane = threadIdx.x & 31;
  const int wave = threadIdx.x >> 5;
  const int hi16 = (lane >> 4) & 1;     // 0: lanes 0-15, 1: lanes 16-31
  const int m    = lane & 15;           // tile row (edge) / B column

  // B fragments in registers. B layout: lanes 0-15 hold K {0..7,16..23};
  // lanes 16-31 hold K {8..15,24..31} (offset by ks*32).
  v16h w1f[4][2], w2f[4][2];
  #pragma unroll
  for (int nt = 0; nt < 4; ++nt) {
    const int n = nt * 16 + m;
    #pragma unroll
    for (int ks = 0; ks < 2; ++ks) {
      const int kb = ks * 32 + hi16 * 8;
      w1f[nt][ks] = cat8(*(const v8h*)&W1h[n * 64 + kb],
                         *(const v8h*)&W1h[n * 64 + kb + 16]);
      w2f[nt][ks] = cat8(*(const v8h*)&W2h[n * 64 + kb],
                         *(const v8h*)&W2h[n * 64 + kb + 16]);
    }
  }

  float b1v[4], b2v[4];
  #pragma unroll
  for (int nt = 0; nt < 4; ++nt) {
    b1v[nt] = b1[nt * 16 + m];
    b2v[nt] = b2[nt * 16 + m];
  }

  const long long eTot     = nEdges + (long long)nNodes;
  const long long numTiles = (eTot + 15) >> 4;           // 106,250 for this problem
  const long long stride   = (long long)gridDim.x * WPB;
  _Float16* hs = hstage[wave];
  int*      ds = dstStage[wave];

  for (long long tile = (long long)blockIdx.x * WPB + wave; tile < numTiles;
       tile += stride) {
    long long e = tile * 16 + m;
    if (e >= eTot) e = eTot - 1;        // duplicate last edge: no-op under max
    int src, dst;
    if (e < nEdges) { src = (int)ei[e]; dst = (int)ei[nEdges + e]; }
    else            { src = dst = (int)(e - nEdges); }   // self loop
    if (!hi16) ds[m] = dst;

    // Prefetch next tile's edge indices (speculative, dropped if OOB).
    {
      long long nt2 = tile + stride;
      if (nt2 < numTiles) __builtin_prefetch(ei + nt2 * 16 + m, 0, 1);
    }

    // ---- A fragment, K-step 0: x[src] features (K = 0..31) ----
    // A layout: lanes 0-15 hold K {0..7,16..23}; lanes 16-31 K {8..15,24..31}.
    const float* xs = x + (long long)src * D_FEAT;
    const int f0 = hi16 * 8;
    f4 xa = *(const f4*)(xs + f0);
    f4 xb = *(const f4*)(xs + f0 + 4);
    f4 xc = *(const f4*)(xs + f0 + 16);
    f4 xd = *(const f4*)(xs + f0 + 20);
    v16h a0;
    #pragma unroll
    for (int i = 0; i < 4; ++i) {
      a0[i]      = (_Float16)xa[i];
      a0[4 + i]  = (_Float16)xb[i];
      a0[8 + i]  = (_Float16)xc[i];
      a0[12 + i] = (_Float16)xd[i];
    }

    // ---- A fragment, K-step 1: dpos in K=32..34, rest zero ----
    v16h a1 = {};
    if (!hi16) {
      const float* ps = pos + (long long)src * 3;
      const float* pd = pos + (long long)dst * 3;
      a1[0] = (_Float16)(ps[0] - pd[0]);
      a1[1] = (_Float16)(ps[1] - pd[1]);
      a1[2] = (_Float16)(ps[2] - pd[2]);
    }

    // ---- Layer 1: h = relu(msg @ W1 + b1), stage row-major f16 to LDS ----
    #pragma unroll
    for (int nt = 0; nt < 4; ++nt) {
      v8f acc;
      #pragma unroll
      for (int r = 0; r < 8; ++r) acc[r] = b1v[nt];
      acc = __builtin_amdgcn_wmma_f32_16x16x32_f16(false, a0, false, w1f[nt][0],
                                                   (short)0, acc, false, false);
      acc = __builtin_amdgcn_wmma_f32_16x16x32_f16(false, a1, false, w1f[nt][1],
                                                   (short)0, acc, false, false);
      const int n = nt * 16 + m;        // D layout: lane holds column n
      #pragma unroll
      for (int r = 0; r < 8; ++r) {     // rows M = r + hi16*8
        float hv = fmaxf(acc[r], 0.0f);
        hs[(r + hi16 * 8) * D_HID + n] = (_Float16)hv;
      }
    }

    // ---- Re-read h as layer-2 A fragments (wave-private LDS transpose) ----
    const _Float16* hrow = hs + m * D_HID;
    const int coff = hi16 * 8;
    v16h h0 = cat8(*(const v8h*)(hrow + coff),
                   *(const v8h*)(hrow + coff + 16));
    v16h h1 = cat8(*(const v8h*)(hrow + 32 + coff),
                   *(const v8h*)(hrow + 32 + coff + 16));

    // dst node index for each of this lane's 8 output rows
    int dr[8];
    #pragma unroll
    for (int r = 0; r < 8; ++r) dr[r] = ds[coff + r];

    // ---- Layer 2 + scatter-max ----
    #pragma unroll
    for (int nt = 0; nt < 4; ++nt) {
      v8f acc;
      #pragma unroll
      for (int r = 0; r < 8; ++r) acc[r] = b2v[nt];
      acc = __builtin_amdgcn_wmma_f32_16x16x32_f16(false, h0, false, w2f[nt][0],
                                                   (short)0, acc, false, false);
      acc = __builtin_amdgcn_wmma_f32_16x16x32_f16(false, h1, false, w2f[nt][1],
                                                   (short)0, acc, false, false);
      const int n = nt * 16 + m;
      #pragma unroll
      for (int r = 0; r < 8; ++r)
        atomicMaxF32(out + (long long)dr[r] * D_OUT + n, acc[r]);
    }
  }
}

extern "C" void kernel_launch(void* const* d_in, const int* in_sizes, int n_in,
                              void* d_out, int out_size, void* d_ws, size_t ws_size,
                              hipStream_t stream) {
  (void)n_in; (void)d_ws; (void)ws_size;
  const float*     x   = (const float*)d_in[0];
  const float*     pos = (const float*)d_in[1];
  const long long* ei  = (const long long*)d_in[2];   // int64 edge_index [2][E]
  const float*     W1  = (const float*)d_in[3];
  const float*     b1  = (const float*)d_in[4];
  const float*     W2  = (const float*)d_in[5];
  const float*     b2  = (const float*)d_in[6];
  float* out = (float*)d_out;

  const int       nNodes = in_sizes[1] / 3;            // pos is N x 3
  const long long nEdges = (long long)(in_sizes[2] / 2);

  init_out_kernel<<<(out_size + BLOCK - 1) / BLOCK, BLOCK, 0, stream>>>(out, out_size);
  pointconv_kernel<<<1024, BLOCK, 0, stream>>>(x, pos, ei, W1, b1, W2, b2,
                                               out, nNodes, nEdges);
}
